// SelfAttention_73014444032668
// MI455X (gfx1250) — matrix operand order repaired and checked by reading, not compiled, not run on hardware
//
#include <hip/hip_runtime.h>
#include <cstdint>

#define DEV __device__ __forceinline__

typedef unsigned short u16;
typedef __attribute__((ext_vector_type(16))) __bf16 v16bf;
typedef __attribute__((ext_vector_type(8)))  float  v8f;
typedef int v4i_g __attribute__((vector_size(16)));  // matches builtin's int4 pointee

static constexpr int Bsz = 4;
static constexpr int Ssz = 2048;
static constexpr int Dsz = 1024;

// --- CDNA5 async Global->LDS copy path (guarded; falls back to sync staging) ---
#if defined(__has_builtin)
#  if __has_builtin(__builtin_amdgcn_global_load_async_to_lds_b128) && \
      __has_builtin(__builtin_amdgcn_s_wait_asynccnt)
#    define USE_ASYNC_LDS 1
#  endif
#endif
#ifndef USE_ASYNC_LDS
#  define USE_ASYNC_LDS 0
#endif

#define GLOBAL_AS __attribute__((address_space(1)))
#define LDS_AS    __attribute__((address_space(3)))

#if USE_ASYNC_LDS
DEV void async_cp16(u16* lds_dst, const u16* gsrc) {
  // GLOBAL_LOAD_ASYNC_TO_LDS_B128: per-lane 16B, LDS written directly (ASYNCcnt).
  // Integer round-trips: generic->AS1 is bit-identical; generic->AS3 is low 32 bits.
  GLOBAL_AS v4i_g* g = (GLOBAL_AS v4i_g*)(uintptr_t)gsrc;
  LDS_AS v4i_g*    l = (LDS_AS v4i_g*)(uint32_t)(uintptr_t)lds_dst;
  __builtin_amdgcn_global_load_async_to_lds_b128(g, l, 0, 0);
}
#endif

// fp32 -> bf16, round-to-nearest-even
DEV u16 f2bf(float f) {
  union { float f; uint32_t u; } v; v.f = f;
  uint32_t r = v.u + 0x7FFFu + ((v.u >> 16) & 1u);
  return (u16)(r >> 16);
}

// load a 32-byte WMMA bf16 fragment as two 16B chunks (works for global and LDS)
DEV v16bf ld_frag(const u16* p0, const u16* p1) {
  union { v16bf v; uint4 q[2]; } f;
  f.q[0] = *(const uint4*)p0;
  f.q[1] = *(const uint4*)p1;
  return f.v;
}

DEV v8f wmma_bf16(v16bf a, v16bf b, v8f c) {
  // (neg_a, A, neg_b, B, c_mod, C, reuse_a, reuse_b)
  return __builtin_amdgcn_wmma_f32_16x16x32_bf16(false, a, false, b, (short)0, c,
                                                 false, false);
}

// ---------------------------------------------------------------------------
// Kernel 1: x fp32 -> bf16 (vectorized)
// ---------------------------------------------------------------------------
__global__ void convert_x_kernel(const float* __restrict__ x, u16* __restrict__ xb, int n4) {
  int i = blockIdx.x * blockDim.x + threadIdx.x;
  if (i < n4) {
    const float4 v = ((const float4*)x)[i];
    uint2 p;
    p.x = (uint32_t)f2bf(v.x) | ((uint32_t)f2bf(v.y) << 16);
    p.y = (uint32_t)f2bf(v.z) | ((uint32_t)f2bf(v.w) << 16);
    ((uint2*)xb)[i] = p;
  }
}

// ---------------------------------------------------------------------------
// Kernel 2: W (D x D, fp32) -> W^T (bf16), tiled through LDS
// ---------------------------------------------------------------------------
__global__ void transpose_w_kernel(const float* __restrict__ Wq, const float* __restrict__ Wk,
                                   const float* __restrict__ Wv, u16* __restrict__ Wt) {
  __shared__ float tile[32][33];
  const float* W = (blockIdx.z == 0) ? Wq : (blockIdx.z == 1) ? Wk : Wv;
  u16* T = Wt + (size_t)blockIdx.z * Dsz * Dsz;
  const int e = blockIdx.x * 32 + threadIdx.x;
  const int d = blockIdx.y * 32 + threadIdx.y;
#pragma unroll
  for (int i = 0; i < 4; ++i)
    tile[threadIdx.y + 8 * i][threadIdx.x] = W[(size_t)(d + 8 * i) * Dsz + e];
  __syncthreads();
  const int dd = blockIdx.y * 32 + threadIdx.x;
  const int ee = blockIdx.x * 32 + threadIdx.y;
#pragma unroll
  for (int i = 0; i < 4; ++i)
    T[(size_t)(ee + 8 * i) * Dsz + dd] = f2bf(tile[threadIdx.x][threadIdx.y + 8 * i]);
}

// ---------------------------------------------------------------------------
// Kernel 3: QKV projection GEMM, 128x128 tile/WG, 8 waves (2x4), 4x2 WMMA/wave.
//   Double-buffered LDS pipeline; tiles stream in via async Global->LDS DMA
//   (GLOBAL_LOAD_ASYNC_TO_LDS_B128 + s_wait_asynccnt) when available.
//   outputs: Q,K row-major bf16; V transposed [b][e][t] bf16
// ---------------------------------------------------------------------------
__global__ __launch_bounds__(256) void qkv_gemm_kernel(
    const u16* __restrict__ xb, const u16* __restrict__ Wt,
    const float* __restrict__ bq, const float* __restrict__ bk, const float* __restrict__ bv,
    u16* __restrict__ Qb, u16* __restrict__ Kb, u16* __restrict__ Vt) {
  // padded rows: stride 80B -> conflict-free ds_load_b128 fragment reads
  __shared__ u16 lx[2][128][40];  // x tile [m][k]
  __shared__ u16 lw[2][128][40];  // W^T tile [n][k]

  const int w = blockIdx.z;
  const u16* Wp = Wt + (size_t)w * Dsz * Dsz;
  const float* bias = (w == 0) ? bq : (w == 1) ? bk : bv;
  const int n0 = blockIdx.x * 128;
  const int m0 = blockIdx.y * 128;
  const int tid = threadIdx.x;
  const int lane = tid & 31, wv = tid >> 5;
  const int wrow = wv >> 2, wcol = wv & 3;
  const int ln16 = lane & 15, lh = lane >> 4;

  const v8f vzero = {0.f, 0.f, 0.f, 0.f, 0.f, 0.f, 0.f, 0.f};
  v8f acc[4][2];
#pragma unroll
  for (int mt = 0; mt < 4; ++mt)
#pragma unroll
    for (int nt = 0; nt < 2; ++nt) acc[mt][nt] = vzero;

  const int c0 = tid * 2;  // two 16B chunks of each tile per thread

  auto stage = [&](int buf, int k0) {
#pragma unroll
    for (int ci = 0; ci < 2; ++ci) {
      const int cc = c0 + ci;
      const int row = cc >> 2, part = cc & 3;
      const u16* gx = &xb[(size_t)(m0 + row) * Dsz + k0 + part * 8];
      const u16* gw = &Wp[(size_t)(n0 + row) * Dsz + k0 + part * 8];
      u16* dx = &lx[buf][row][part * 8];
      u16* dw = &lw[buf][row][part * 8];
#if USE_ASYNC_LDS
      async_cp16(dx, gx);
      async_cp16(dw, gw);
#else
      *(uint4*)dx = *(const uint4*)gx;
      *(uint4*)dw = *(const uint4*)gw;
#endif
    }
  };

  stage(0, 0);  // prologue: tile 0 in flight

  for (int kk = 0; kk < 32; ++kk) {
    const int cur = kk & 1;
#if USE_ASYNC_LDS
    __builtin_amdgcn_s_wait_asynccnt(0);  // my tile-kk copies have landed
#endif
    // barrier: everyone's tile-kk data is in LDS, and everyone is done
    // reading buffer (1-cur) from iteration kk-1 -> safe to overwrite it.
    __syncthreads();
    if (kk < 31) stage(1 - cur, (kk + 1) * 32);

    v16bf afr[4], bfr[2];
#pragma unroll
    for (int mt = 0; mt < 4; ++mt) {
      const u16* r = &lx[cur][wrow * 64 + mt * 16 + ln16][0];
      afr[mt] = ld_frag(r + 8 * lh, r + 16 + 8 * lh);
    }
#pragma unroll
    for (int nt = 0; nt < 2; ++nt) {
      const u16* r = &lw[cur][wcol * 32 + nt * 16 + ln16][0];
      bfr[nt] = ld_frag(r + 8 * lh, r + 16 + 8 * lh);
    }
#pragma unroll
    for (int mt = 0; mt < 4; ++mt)
#pragma unroll
      for (int nt = 0; nt < 2; ++nt)
        acc[mt][nt] = wmma_bf16(afr[mt], bfr[nt], acc[mt][nt]);
  }

  // epilogue: bias add, convert, store
#pragma unroll
  for (int nt = 0; nt < 2; ++nt) {
    const int n = n0 + wcol * 32 + nt * 16 + ln16;
    const float bb = bias[n];
#pragma unroll
    for (int mt = 0; mt < 4; ++mt) {
#pragma unroll
      for (int r = 0; r < 8; ++r) {
        const int m = m0 + wrow * 64 + mt * 16 + 8 * lh + r;
        const u16 h = f2bf(acc[mt][nt][r] + bb);
        if (w == 0)      Qb[(size_t)m * Dsz + n] = h;
        else if (w == 1) Kb[(size_t)m * Dsz + n] = h;
        else {
          const int bidx = m >> 11, s = m & 2047;
          Vt[(size_t)bidx * Dsz * Ssz + (size_t)n * Ssz + s] = h;
        }
      }
    }
  }
}

// ---------------------------------------------------------------------------
// Kernel 4: flash-style attention. 32 Q-rows per WG, stream T in 256-col tiles.
//   Phase A: scores (WMMA, frags straight from L2-resident Q/K)
//   Phase B: online softmax (shuffle reductions + LDS row stats), P -> bf16 LDS
//   Phase C: O = alpha*O + P@V (wave owns a 128-wide D slice; V^T B-frags)
// ---------------------------------------------------------------------------
__global__ __launch_bounds__(256) void attn_kernel(const u16* __restrict__ Qb,
                                                   const u16* __restrict__ Kb,
                                                   const u16* __restrict__ Vt,
                                                   float* __restrict__ out) {
  __shared__ float lsc[32][257];   // raw scores for current T tile
  __shared__ u16   lp[32][264];    // P (bf16) for current T tile
  __shared__ float lmx[32], lsum[32], lalpha[32];

  const int b = blockIdx.y;
  const int s0 = blockIdx.x * 32;
  const u16* Qp = Qb + ((size_t)b * Ssz + s0) * Dsz;
  const u16* Kp = Kb + (size_t)b * Ssz * Dsz;
  const u16* Vp = Vt + (size_t)b * Dsz * Ssz;
  float* Op = out + ((size_t)b * Ssz + s0) * Dsz;

  const int tid = threadIdx.x;
  const int lane = tid & 31, wv = tid >> 5;
  const int ln16 = lane & 15, lh = lane >> 4;

  if (tid < 32) { lmx[tid] = -3.0e38f; lsum[tid] = 0.0f; }

  const v8f vzero = {0.f, 0.f, 0.f, 0.f, 0.f, 0.f, 0.f, 0.f};
  v8f oacc[2][8];
#pragma unroll
  for (int mt = 0; mt < 2; ++mt)
#pragma unroll
    for (int nt = 0; nt < 8; ++nt) oacc[mt][nt] = vzero;

  __syncthreads();

  for (int tt = 0; tt < 8; ++tt) {
    const int t0 = tt * 256;

    // ---- Phase A: scores [32 x 256]; wave wv computes t-cols [wv*32, +32) ----
    v8f sacc[2][2];
#pragma unroll
    for (int mt = 0; mt < 2; ++mt)
#pragma unroll
      for (int nt = 0; nt < 2; ++nt) sacc[mt][nt] = vzero;

    for (int kk = 0; kk < 32; ++kk) {
      const int k = kk * 32;
      v16bf af[2], bk2[2];
#pragma unroll
      for (int mt = 0; mt < 2; ++mt) {
        const u16* r = Qp + (size_t)(mt * 16 + ln16) * Dsz + k;
        af[mt] = ld_frag(r + 8 * lh, r + 16 + 8 * lh);
      }
#pragma unroll
      for (int nt = 0; nt < 2; ++nt) {
        const int t = t0 + wv * 32 + nt * 16 + ln16;
        const u16* r = Kp + (size_t)t * Dsz + k;
        bk2[nt] = ld_frag(r + 8 * lh, r + 16 + 8 * lh);
      }
#pragma unroll
      for (int mt = 0; mt < 2; ++mt)
#pragma unroll
        for (int nt = 0; nt < 2; ++nt)
          sacc[mt][nt] = wmma_bf16(af[mt], bk2[nt], sacc[mt][nt]);
    }
#pragma unroll
    for (int mt = 0; mt < 2; ++mt)
#pragma unroll
      for (int nt = 0; nt < 2; ++nt)
#pragma unroll
        for (int r = 0; r < 8; ++r) {
          const int row = mt * 16 + 8 * lh + r;
          const int col = wv * 32 + nt * 16 + ln16;
          lsc[row][col] = sacc[mt][nt][r] * 0.03125f;  // 1/sqrt(1024)
        }
    __syncthreads();

    // ---- Phase B: online softmax; wave wv owns rows [4wv, 4wv+4) ----
    for (int ri = 0; ri < 4; ++ri) {
      const int row = wv * 4 + ri;
      float vals[8];
      float vmax = -3.0e38f;
#pragma unroll
      for (int j = 0; j < 8; ++j) {
        vals[j] = lsc[row][lane + 32 * j];
        vmax = fmaxf(vmax, vals[j]);
      }
#pragma unroll
      for (int off = 16; off >= 1; off >>= 1)
        vmax = fmaxf(vmax, __shfl_xor(vmax, off, 32));
      const float mold = lmx[row];
      const float mnew = fmaxf(mold, vmax);
      float psum = 0.f;
#pragma unroll
      for (int j = 0; j < 8; ++j) {
        const float p = __expf(vals[j] - mnew);
        psum += p;
        lp[row][lane + 32 * j] = f2bf(p);
      }
#pragma unroll
      for (int off = 16; off >= 1; off >>= 1)
        psum += __shfl_xor(psum, off, 32);
      if (lane == 0) {
        const float alpha = __expf(mold - mnew);
        lsum[row] = lsum[row] * alpha + psum;
        lmx[row] = mnew;
        lalpha[row] = alpha;
      }
    }
    __syncthreads();

    // ---- Phase C: O = alpha*O + P @ V; wave owns D cols [wv*128, +128) ----
    float arow[2][8];
#pragma unroll
    for (int mt = 0; mt < 2; ++mt)
#pragma unroll
      for (int r = 0; r < 8; ++r) arow[mt][r] = lalpha[mt * 16 + 8 * lh + r];
#pragma unroll
    for (int mt = 0; mt < 2; ++mt)
#pragma unroll
      for (int nt = 0; nt < 8; ++nt)
#pragma unroll
        for (int r = 0; r < 8; ++r) oacc[mt][nt][r] *= arow[mt][r];

    for (int kk = 0; kk < 8; ++kk) {
      const int kt = kk * 32;
      v16bf pf[2];
#pragma unroll
      for (int mt = 0; mt < 2; ++mt) {
        const u16* r = &lp[mt * 16 + ln16][kt];
        pf[mt] = ld_frag(r + 8 * lh, r + 16 + 8 * lh);
      }
#pragma unroll
      for (int nt = 0; nt < 8; ++nt) {
        const int e = wv * 128 + nt * 16 + ln16;
        const u16* r = Vp + (size_t)e * Ssz + t0 + kt;
        const v16bf vf = ld_frag(r + 8 * lh, r + 16 + 8 * lh);
#pragma unroll
        for (int mt = 0; mt < 2; ++mt)
          oacc[mt][nt] = wmma_bf16(pf[mt], vf, oacc[mt][nt]);
      }
    }
    __syncthreads();
  }

  // ---- epilogue: divide by row sums, store fp32 ----
  float linv[2][8];
#pragma unroll
  for (int mt = 0; mt < 2; ++mt)
#pragma unroll
    for (int r = 0; r < 8; ++r) linv[mt][r] = 1.0f / lsum[mt * 16 + 8 * lh + r];
#pragma unroll
  for (int mt = 0; mt < 2; ++mt)
#pragma unroll
    for (int nt = 0; nt < 8; ++nt)
#pragma unroll
      for (int r = 0; r < 8; ++r) {
        const int row = mt * 16 + 8 * lh + r;
        const int col = wv * 128 + nt * 16 + ln16;
        Op[(size_t)row * Dsz + col] = oacc[mt][nt][r] * linv[mt][r];
      }
}

// ---------------------------------------------------------------------------
extern "C" void kernel_launch(void* const* d_in, const int* in_sizes, int n_in,
                              void* d_out, int out_size, void* d_ws, size_t ws_size,
                              hipStream_t stream) {
  const float* x  = (const float*)d_in[0];
  const float* Wq = (const float*)d_in[1];
  const float* bq = (const float*)d_in[2];
  const float* Wk = (const float*)d_in[3];
  const float* bk = (const float*)d_in[4];
  const float* Wv = (const float*)d_in[5];
  const float* bv = (const float*)d_in[6];
  float* out = (float*)d_out;

  // workspace layout (bf16): xb | Wt(x3) | Q | K | V^T   (~73 MB)
  u16* xb = (u16*)d_ws;
  u16* Wt = xb + (size_t)Bsz * Ssz * Dsz;
  u16* Qb = Wt + (size_t)3 * Dsz * Dsz;
  u16* Kb = Qb + (size_t)Bsz * Ssz * Dsz;
  u16* Vt = Kb + (size_t)Bsz * Ssz * Dsz;

  const int n4 = Bsz * Ssz * Dsz / 4;
  convert_x_kernel<<<n4 / 256, 256, 0, stream>>>(x, xb, n4);
  transpose_w_kernel<<<dim3(Dsz / 32, Dsz / 32, 3), dim3(32, 8), 0, stream>>>(Wq, Wk, Wv, Wt);
  qkv_gemm_kernel<<<dim3(Dsz / 128, (Bsz * Ssz) / 128, 3), 256, 0, stream>>>(
      xb, Wt, bq, bk, bv, Qb, Kb, Vt);
  attn_kernel<<<dim3(Ssz / 32, Bsz), 256, 0, stream>>>(Qb, Kb, Vt, out);
}
